// WordEmbeddingAdapter_41944650613067
// MI455X (gfx1250) — matrix operand order repaired and checked by reading, not compiled, not run on hardware
//
#include <hip/hip_runtime.h>
#include <hip/hip_bf16.h>
#include <math.h>

typedef __attribute__((ext_vector_type(16))) _Float16 v16h;
typedef __attribute__((ext_vector_type(8)))  _Float16 v8h;
typedef __attribute__((ext_vector_type(8)))  float    v8f;
typedef __attribute__((ext_vector_type(4)))  int      v4i;

#define AS1 __attribute__((address_space(1)))
#define AS3 __attribute__((address_space(3)))

#if __has_builtin(__builtin_amdgcn_global_load_async_to_lds_b128)
#define HAVE_ASYNC_LDS 1
#else
#define HAVE_ASYNC_LDS 0
#endif

#define HID     768
#define NW      5
#define DW      200
#define DWP     224      // K=200 padded to 7*32
#define ROWS_BL 8192     // B*L
#define ROWS_BLN 40960   // B*L*N
#define LN_EPS  1e-12f

// ---------------------------------------------------------------------------
// f32 -> f16 conversion with 2-D zero padding (row-major, for A operands)
// ---------------------------------------------------------------------------
__global__ void cvt_pad_f32_f16(const float* __restrict__ in, _Float16* __restrict__ out,
                                int rin, int cin, int rout, int cout) {
    size_t i = (size_t)blockIdx.x * blockDim.x + threadIdx.x;
    size_t total = (size_t)rout * cout;
    if (i >= total) return;
    int c = (int)(i % cout);
    size_t r = i / cout;
    float v = 0.0f;
    if (c < cin && r < (size_t)rin) v = in[r * cin + c];
    out[i] = (_Float16)v;
}

__global__ void cvt_f32_f16(const float* __restrict__ in, _Float16* __restrict__ out, size_t n) {
    size_t i = (size_t)blockIdx.x * blockDim.x + threadIdx.x;
    if (i < n) out[i] = (_Float16)in[i];
}

// Transpose + K-pad: in[kin][ncols] f32  ->  out[ncols][kout] f16 (zeros for k>=kin)
__global__ void cvt_t_pad_f32_f16(const float* __restrict__ in, _Float16* __restrict__ out,
                                  int kin, int ncols, int kout) {
    size_t i = (size_t)blockIdx.x * blockDim.x + threadIdx.x;
    size_t total = (size_t)ncols * kout;
    if (i >= total) return;
    int k = (int)(i % kout);
    size_t n = i / kout;
    float v = (k < kin) ? in[(size_t)k * ncols + n] : 0.0f;
    out[i] = (_Float16)v;
}

// ---------------------------------------------------------------------------
// 16-byte global -> LDS staging chunk: async-to-LDS when available.
// ---------------------------------------------------------------------------
__device__ __forceinline__ void stage16(const _Float16* g, _Float16* l) {
#if HAVE_ASYNC_LDS
    __builtin_amdgcn_global_load_async_to_lds_b128((AS1 v4i*)g, (AS3 v4i*)l, 0, 0);
#else
    *(v8h*)l = *(const v8h*)g;
#endif
}

__device__ __forceinline__ void stage_wait() {
#if HAVE_ASYNC_LDS
#if __has_builtin(__builtin_amdgcn_s_wait_asynccnt)
    __builtin_amdgcn_s_wait_asynccnt(0);
#else
    asm volatile("s_wait_asynccnt 0x0" ::: "memory");
#endif
#endif
}

// ---------------------------------------------------------------------------
// Tiled WMMA GEMM: C[M,N] = epilogue(A[M,K] @ Bt[N,K]^T + bias)
//   A row-major [M,K] f16; Bt TRANSPOSED weights [N,K] f16 (row n = column n of B).
//   K % 32 == 0, M % 128 == 0, N % 64 == 0.
//   MODE 0: C f32, no bias.  MODE 1: C f16 = tanh(acc+bias).  MODE 2: C f16 = acc+bias.
// Block: 256 threads (8 waves). Block tile 128x64; each wave owns a 32x32 subtile
// (2x2 v_wmma_f32_16x16x32_f16 accumulators).
// ---------------------------------------------------------------------------
template <int MODE>
__global__ __launch_bounds__(256)
void gemm_wmma_f16(const _Float16* __restrict__ A, const _Float16* __restrict__ Bt,
                   const float* __restrict__ bias, void* __restrict__ C,
                   int M, int N, int K) {
    // +8 halves of padding -> 80-byte row stride (16B aligned, conflict-mitigating)
    __shared__ __align__(16) _Float16 As[128][40];
    __shared__ __align__(16) _Float16 Bs[64][40];

    const int tid  = threadIdx.x;
    const int lane = tid & 31;
    const int w    = tid >> 5;
    const int wm   = w & 3;        // 4 waves along M (32 rows each)
    const int wn   = w >> 2;       // 2 waves along N (32 cols each)
    const int mBlk = blockIdx.y * 128;
    const int nBlk = blockIdx.x * 64;

    const int g  = lane >> 4;      // K-half group held by this lane
    const int lm = lane & 15;

    // per-thread staging coordinates
    const int arow0 = tid >> 2;                 // A chunk 0 row (0..63)
    const int akoff = (tid & 3) << 3;           // A chunk k offset (0,8,16,24)
    const int brow  = tid >> 2;                 // B row n (0..63)
    const int bkoff = (tid & 3) << 3;

    v8f acc[2][2] = {};

    for (int k0 = 0; k0 < K; k0 += 32) {
        // ---- stage A tile 128x32: 512 x 16B chunks, 2 per thread ----
        stage16(A + (size_t)(mBlk + arow0) * K + k0 + akoff, &As[arow0][akoff]);
        stage16(A + (size_t)(mBlk + arow0 + 64) * K + k0 + akoff, &As[arow0 + 64][akoff]);
        // ---- stage B tile 64x32 (already transposed in global): 1 chunk/thread ----
        stage16(Bt + (size_t)(nBlk + brow) * K + k0 + bkoff, &Bs[brow][bkoff]);

        // prefetch next k-tile (global_prefetch_b8)
        if (k0 + 32 < K) {
            __builtin_prefetch(A + (size_t)(mBlk + arow0) * K + k0 + 32 + akoff, 0, 1);
            __builtin_prefetch(Bt + (size_t)(nBlk + brow) * K + k0 + 32 + bkoff, 0, 1);
        }

        stage_wait();
        __syncthreads();

        // ---- load fragments per ISA 16-bit A/B layouts ----
        // A 16x32: lanes 0-15 hold K {0..7,16..23}, lanes 16-31 hold K {8..15,24..31}
        v16h af[2], bf[2];
        #pragma unroll
        for (int i = 0; i < 2; ++i) {
            int m = wm * 32 + i * 16 + lm;
            v8h lo = *(const v8h*)&As[m][g * 8];
            v8h hi = *(const v8h*)&As[m][16 + g * 8];
            af[i] = __builtin_shufflevector(lo, hi, 0,1,2,3,4,5,6,7,8,9,10,11,12,13,14,15);
        }
        // B 32x16: lanes 0-15 hold K {0..7,16..23} of column n, lanes 16-31 hold K {8..15,24..31}
        #pragma unroll
        for (int j = 0; j < 2; ++j) {
            int n = wn * 32 + j * 16 + lm;
            v8h lo = *(const v8h*)&Bs[n][g * 8];
            v8h hi = *(const v8h*)&Bs[n][16 + g * 8];
            bf[j] = __builtin_shufflevector(lo, hi, 0,1,2,3,4,5,6,7,8,9,10,11,12,13,14,15);
        }
        #pragma unroll
        for (int i = 0; i < 2; ++i)
            #pragma unroll
            for (int j = 0; j < 2; ++j)
                acc[i][j] = __builtin_amdgcn_wmma_f32_16x16x32_f16(
                    false, af[i], false, bf[j], (short)0, acc[i][j], false, false);
        __syncthreads();
    }

    // ---- epilogue: C/D layout -> lane: M = r + 8*(lane/16), N = lane%16 ----
    #pragma unroll
    for (int i = 0; i < 2; ++i) {
        #pragma unroll
        for (int j = 0; j < 2; ++j) {
            int gn = nBlk + wn * 32 + j * 16 + lm;
            float bv = 0.0f;
            if (MODE != 0) bv = bias[gn];
            #pragma unroll
            for (int r = 0; r < 8; ++r) {
                int gm = mBlk + wm * 32 + i * 16 + r + (g << 3);
                float v = acc[i][j][r] + bv;
                if (MODE == 1) v = tanhf(v);
                if (MODE == 0) ((float*)C)[(size_t)gm * N + gn] = v;
                else           ((_Float16*)C)[(size_t)gm * N + gn] = (_Float16)v;
            }
        }
    }
}

// ---------------------------------------------------------------------------
// Fused tail: scores = proj . WO_n, mask, softmax(5), weighted sum, residual, LN
// One wave (32 lanes) per (b,l) row; 768/32 = 24 elements per lane.
// ---------------------------------------------------------------------------
__global__ __launch_bounds__(256)
void fuse_attn_ln(const float* __restrict__ proj, const _Float16* __restrict__ wo,
                  const float* __restrict__ lo, const int* __restrict__ mask,
                  const float* __restrict__ gamma, const float* __restrict__ beta,
                  float* __restrict__ out) {
    const int lane = threadIdx.x & 31;
    const int row  = blockIdx.x * (blockDim.x >> 5) + (threadIdx.x >> 5);
    const float*    prow = proj + (size_t)row * HID;
    const float*    lrow = lo   + (size_t)row * HID;
    const _Float16* wrow = wo   + (size_t)row * NW * HID;

    // 5 dot products of length 768
    float s[NW] = {0.f, 0.f, 0.f, 0.f, 0.f};
    #pragma unroll
    for (int t = 0; t < HID / 32; ++t) {
        int j = lane + t * 32;
        float p = prow[j];
        #pragma unroll
        for (int n = 0; n < NW; ++n) s[n] += p * (float)wrow[n * HID + j];
    }
    #pragma unroll
    for (int n = 0; n < NW; ++n)
        for (int off = 16; off > 0; off >>= 1) s[n] += __shfl_xor(s[n], off, 32);

    // mask + softmax over NW=5
    float mx = -1e30f;
    #pragma unroll
    for (int n = 0; n < NW; ++n) {
        if (mask[row * NW + n] != 0) s[n] = -1e9f;
        mx = fmaxf(mx, s[n]);
    }
    float denom = 0.f;
    #pragma unroll
    for (int n = 0; n < NW; ++n) { s[n] = expf(s[n] - mx); denom += s[n]; }
    float inv = 1.f / denom;
    #pragma unroll
    for (int n = 0; n < NW; ++n) s[n] *= inv;

    // weighted sum + residual
    float o[HID / 32];
    float mean = 0.f;
    #pragma unroll
    for (int t = 0; t < HID / 32; ++t) {
        int j = lane + t * 32;
        float acc = lrow[j];
        #pragma unroll
        for (int n = 0; n < NW; ++n) acc += s[n] * (float)wrow[n * HID + j];
        o[t] = acc;
        mean += acc;
    }
    for (int off = 16; off > 0; off >>= 1) mean += __shfl_xor(mean, off, 32);
    mean *= (1.0f / HID);
    float var = 0.f;
    #pragma unroll
    for (int t = 0; t < HID / 32; ++t) { float d = o[t] - mean; var += d * d; }
    for (int off = 16; off > 0; off >>= 1) var += __shfl_xor(var, off, 32);
    var *= (1.0f / HID);
    float rs = rsqrtf(var + LN_EPS);
    #pragma unroll
    for (int t = 0; t < HID / 32; ++t) {
        int j = lane + t * 32;
        out[(size_t)row * HID + j] = (o[t] - mean) * rs * gamma[j] + beta[j];
    }
}

// ---------------------------------------------------------------------------
extern "C" void kernel_launch(void* const* d_in, const int* in_sizes, int n_in,
                              void* d_out, int out_size, void* d_ws, size_t ws_size,
                              hipStream_t stream) {
    const float* LO    = (const float*)d_in[0];   // [B,L,HID]
    const float* WE    = (const float*)d_in[1];   // [B,L,N,DW]
    const int*   WMASK = (const int*)d_in[2];     // [B,L,N]
    const float* W1    = (const float*)d_in[3];   // [DW,HID]
    const float* b1    = (const float*)d_in[4];   // [HID]
    const float* W2    = (const float*)d_in[5];   // [HID,HID]
    const float* b2    = (const float*)d_in[6];   // [HID]
    const float* AW    = (const float*)d_in[7];   // [HID,HID]
    const float* gamma = (const float*)d_in[8];
    const float* beta  = (const float*)d_in[9];
    float* out = (float*)d_out;

    char* ws = (char*)d_ws;
    size_t off = 0;
    auto alloc = [&](size_t bytes) -> char* {
        char* p = ws + off;
        off += (bytes + 255) & ~(size_t)255;
        return p;
    };

    _Float16* A1h  = (_Float16*)alloc((size_t)ROWS_BLN * DWP * 2);  // WE f16, K-padded
    _Float16* W1t  = (_Float16*)alloc((size_t)HID * DWP * 2);       // W1^T f16 [768,224]
    _Float16* X1h  = (_Float16*)alloc((size_t)ROWS_BLN * HID * 2);  // tanh(WE@W1+b1)
    _Float16* W2t  = (_Float16*)alloc((size_t)HID * HID * 2);       // W2^T
    _Float16* WOh  = (_Float16*)alloc((size_t)ROWS_BLN * HID * 2);  // word_outputs
    _Float16* LOh  = (_Float16*)alloc((size_t)ROWS_BL * HID * 2);
    _Float16* AWt  = (_Float16*)alloc((size_t)HID * HID * 2);       // attn_W^T
    float*    proj = (float*)alloc((size_t)ROWS_BL * HID * 4);      // LO @ attn_W

    // --- conversions (A operands row-major; weights transposed to [N][Kpad]) ---
    {
        size_t n = (size_t)ROWS_BLN * DWP;
        cvt_pad_f32_f16<<<(int)((n + 255) / 256), 256, 0, stream>>>(WE, A1h, ROWS_BLN, DW, ROWS_BLN, DWP);
    }
    {
        size_t n = (size_t)HID * DWP;
        cvt_t_pad_f32_f16<<<(int)((n + 255) / 256), 256, 0, stream>>>(W1, W1t, DW, HID, DWP);
    }
    {
        size_t n = (size_t)HID * HID;
        cvt_t_pad_f32_f16<<<(int)((n + 255) / 256), 256, 0, stream>>>(W2, W2t, HID, HID, HID);
        cvt_t_pad_f32_f16<<<(int)((n + 255) / 256), 256, 0, stream>>>(AW, AWt, HID, HID, HID);
    }
    {
        size_t n = (size_t)ROWS_BL * HID;
        cvt_f32_f16<<<(int)((n + 255) / 256), 256, 0, stream>>>(LO, LOh, n);
    }

    // --- GEMM 1: X1 = tanh(WE @ W1 + b1)   [40960 x 768], K = 224 ---
    gemm_wmma_f16<1><<<dim3(HID / 64, ROWS_BLN / 128), 256, 0, stream>>>(
        A1h, W1t, b1, X1h, ROWS_BLN, HID, DWP);

    // --- GEMM 2: WO = X1 @ W2 + b2         [40960 x 768], K = 768 ---
    gemm_wmma_f16<2><<<dim3(HID / 64, ROWS_BLN / 128), 256, 0, stream>>>(
        X1h, W2t, b2, WOh, ROWS_BLN, HID, HID);

    // --- GEMM 3: proj = LO @ attn_W        [8192 x 768],  K = 768 ---
    gemm_wmma_f16<0><<<dim3(HID / 64, ROWS_BL / 128), 256, 0, stream>>>(
        LOh, AWt, nullptr, proj, ROWS_BL, HID, HID);

    // --- fused scores/softmax/weighted/residual/LayerNorm ---
    fuse_attn_ln<<<ROWS_BL / 8, 256, 0, stream>>>(proj, WOh, LO, WMASK, gamma, beta, out);
}
